// FFNN_26044681682967
// MI455X (gfx1250) — matrix operand order repaired and checked by reading, not compiled, not run on hardware
//
#include <hip/hip_runtime.h>

#define EPSF 1e-15f

typedef __attribute__((ext_vector_type(16))) __bf16 v16bf;
typedef __attribute__((ext_vector_type(8)))  float  v8f;

struct Frag32 { uint4 lo, hi; };

__device__ inline unsigned short f2bf(float f) {
  unsigned int u = __float_as_uint(f);
  unsigned int r = u + 0x7FFFu + ((u >> 16) & 1u);
  return (unsigned short)(r >> 16);
}

__device__ inline float artanh_c(float x) {
  x = fminf(fmaxf(x, -1.0f + 1e-5f), 1.0f - 1e-5f);
  return 0.5f * logf((1.0f + x) / (1.0f - x));
}

// wave32 shuffle reduction + cross-wave LDS combine (2 barriers total)
__device__ inline float block_sum(float v, float* sm) {
  for (int o = 16; o > 0; o >>= 1) v += __shfl_xor(v, o, 32);
  const int t = threadIdx.x;
  const int nw = blockDim.x >> 5;
  if ((t & 31) == 0) sm[t >> 5] = v;
  __syncthreads();
  float r = 0.0f;
  for (int i = 0; i < nw; ++i) r += sm[i];
  __syncthreads();
  return r;
}

__device__ inline float mobius_add_elem(float x, float y, float x2, float y2, float xy) {
  float num = (1.0f + 2.0f * xy + y2) * x + (1.0f - x2) * y;
  float den = fmaxf(1.0f + 2.0f * xy + x2 * y2, EPSF);
  return num / den;
}

// mobius_matvec per-element rescale: raw Mx value -> point on ball
__device__ inline float mmv_scale(float mx, float mxn, float xn) {
  float r = tanhf(mxn / xn * artanh_c(xn)) * mx / mxn;
  return (mxn <= 1e-7f) ? 0.0f : r;
}

// ---------------------------------------------------------------------------
// Async global->LDS copy (CDNA5): 16B per lane, tracked by ASYNCcnt.
// LDS address operand = low 32 bits of the generic pointer (ISA 10.2).
// ---------------------------------------------------------------------------
__device__ inline void async_load_b128(const void* g, void* lds) {
  unsigned int l = (unsigned int)(unsigned long long)lds;
  asm volatile("global_load_async_to_lds_b128 %0, %1, off" :: "v"(l), "v"(g) : "memory");
}
__device__ inline void async_wait0() {
  asm volatile("s_wait_asynccnt 0x0" ::: "memory");
}

// Stage a 64(N) x 32(K) bf16 weight tile into LDS: 256 x 16B chunks, 128 threads.
__device__ inline void stage_B(const unsigned short* __restrict__ W, int ldw,
                               int ntile, int kk, unsigned short* buf, int tid) {
#pragma unroll
  for (int c = tid; c < 256; c += 128) {
    int row = c >> 2;   // 0..63 (n within tile)
    int q   = c & 3;    // 16B quarter of the 64B row
    const unsigned short* g = W + (ntile + row) * ldw + kk + q * 8;
    async_load_b128(g, buf + row * 32 + q * 8);
  }
}

// ---------------------------------------------------------------------------
// GEMM: C[M,N] = A[M,K] * W[N,K]^T  (bf16 in, fp32 accumulate)
// Block = 4 waves = 64(M) x 64(N) tile. B tile double-buffered in LDS via
// async loads; each wave computes a 16x64 slice with v_wmma_f32_16x16x32_bf16.
// ---------------------------------------------------------------------------
__global__ __launch_bounds__(128) void wmma_gemm_kernel(
    const unsigned short* __restrict__ A, int lda,
    const unsigned short* __restrict__ W, int ldw,
    float* __restrict__ C, int ldc, int K)
{
  __shared__ unsigned short Bt[2][64 * 32];
  const int ntile = blockIdx.x * 64;
  const int mtile = blockIdx.y * 64;
  const int tid   = threadIdx.x;
  const int wid   = tid >> 5;
  const int lane  = tid & 31;
  const int r     = lane & 15;
  const int ak0   = (lane >> 4) * 16;  // A frag K sub-offset per lane half
  const int bk0   = (lane >> 4) * 16;  // B frag K sub-offset per lane half

  v8f acc0 = {}, acc1 = {}, acc2 = {}, acc3 = {};
  const unsigned short* Arow = A + (mtile + wid * 16 + r) * lda;

  stage_B(W, ldw, ntile, 0, Bt[0], tid);
  async_wait0();
  __syncthreads();

  for (int kk = 0; kk < K; kk += 32) {
    const int cur = (kk >> 5) & 1;
    if (kk + 32 < K) stage_B(W, ldw, ntile, kk + 32, Bt[cur ^ 1], tid);

    Frag32 fa;
    fa.lo = *(const uint4*)(Arow + kk + ak0);
    fa.hi = *(const uint4*)(Arow + kk + ak0 + 8);
    v16bf a = __builtin_bit_cast(v16bf, fa);

    const unsigned short* bp = Bt[cur] + r * 32 + bk0;
    Frag32 fb;

    fb.lo = *(const uint4*)(bp);
    fb.hi = *(const uint4*)(bp + 8);
    v16bf b0 = __builtin_bit_cast(v16bf, fb);
    acc0 = __builtin_amdgcn_wmma_f32_16x16x32_bf16(false, a, false, b0, (short)0, acc0, false, false);

    fb.lo = *(const uint4*)(bp + 16 * 32);
    fb.hi = *(const uint4*)(bp + 16 * 32 + 8);
    v16bf b1 = __builtin_bit_cast(v16bf, fb);
    acc1 = __builtin_amdgcn_wmma_f32_16x16x32_bf16(false, a, false, b1, (short)0, acc1, false, false);

    fb.lo = *(const uint4*)(bp + 32 * 32);
    fb.hi = *(const uint4*)(bp + 32 * 32 + 8);
    v16bf b2 = __builtin_bit_cast(v16bf, fb);
    acc2 = __builtin_amdgcn_wmma_f32_16x16x32_bf16(false, a, false, b2, (short)0, acc2, false, false);

    fb.lo = *(const uint4*)(bp + 48 * 32);
    fb.hi = *(const uint4*)(bp + 48 * 32 + 8);
    v16bf b3 = __builtin_bit_cast(v16bf, fb);
    acc3 = __builtin_amdgcn_wmma_f32_16x16x32_bf16(false, a, false, b3, (short)0, acc3, false, false);

    async_wait0();     // own async stores into next buffer complete
    __syncthreads();   // all waves done consuming cur + staging next
  }

  const int mbase = (lane >> 4) * 8;
  float* Cr = C + ntile + r;
#pragma unroll
  for (int i = 0; i < 8; ++i) {
    int m = mtile + wid * 16 + mbase + i;
    Cr[m * ldc +  0] = acc0[i];
    Cr[m * ldc + 16] = acc1[i];
    Cr[m * ldc + 32] = acc2[i];
    Cr[m * ldc + 48] = acc3[i];
  }
}

// ---------------------------------------------------------------------------
__global__ void f32_to_bf16_kernel(const float* __restrict__ src,
                                   unsigned short* __restrict__ dst, int n) {
  int i = blockIdx.x * blockDim.x + threadIdx.x;
  if (i < n) dst[i] = f2bf(src[i]);
}

__global__ void init_h_kernel(float* h, unsigned short* h_bf, float* hn, int BH, int B) {
  int i = blockIdx.x * blockDim.x + threadIdx.x;
  if (i < BH) { h[i] = 0.0f; h_bf[i] = 0; }
  if (i < B) hn[i] = 3.16227766e-8f;  // sqrt(EPS), matches _norm of zero vector
}

// x_seq[s,b,:] = expmap0(embedding[tokens[b,s]]); also per-row norm of the point.
__global__ __launch_bounds__(256) void embed_expmap0_kernel(
    const float* __restrict__ emb, const int* __restrict__ tokens,
    unsigned short* __restrict__ x_bf, float* __restrict__ xn_out,
    int B, int S, int E)
{
  extern __shared__ float sm[];
  int sb = blockIdx.x;
  int s = sb / B, b = sb - s * B;
  int t = threadIdx.x;
  int tok = tokens[b * S + s];
  float u = emb[(long)tok * E + t];
  float un = sqrtf(fmaxf(block_sum(u * u, sm), EPSF));
  float x = tanhf(un) * u / un;
  x_bf[(long)sb * E + t] = f2bf(x);
  float xn = sqrtf(fmaxf(block_sum(x * x, sm), EPSF));
  if (t == 0) xn_out[sb] = xn;
}

// ---------------------------------------------------------------------------
// Per-row hyperbolic gate math (one block = one batch row, H threads)
// ---------------------------------------------------------------------------
__device__ inline float gate_elem(float wh, float ux, float bias, float hn, float xn, float* sm) {
  float whn = sqrtf(fmaxf(block_sum(wh * wh, sm), EPSF));
  float a   = mmv_scale(wh, whn, hn);
  float uxn = sqrtf(fmaxf(block_sum(ux * ux, sm), EPSF));
  float bb  = mmv_scale(ux, uxn, xn);
  float x2 = block_sum(a * a, sm);
  float y2 = block_sum(bb * bb, sm);
  float xy = block_sum(a * bb, sm);
  float c = mobius_add_elem(a, bb, x2, y2, xy);
  float c2 = block_sum(c * c, sm);
  float b2 = block_sum(bias * bias, sm);
  float cb = block_sum(c * bias, sm);
  float d = mobius_add_elem(c, bias, c2, b2, cb);
  float dnn = sqrtf(fmaxf(block_sum(d * d, sm), EPSF));
  float lm = artanh_c(dnn) * d / dnn;          // logmap0
  return 1.0f / (1.0f + expf(-lm));            // sigmoid
}

__global__ __launch_bounds__(512) void gru_gates_kernel(
    const float* __restrict__ Whz, const float* __restrict__ Whr, int ldWh,
    const float* __restrict__ Ux, int ldUx,
    const float* __restrict__ h, const float* __restrict__ hn_arr,
    const float* __restrict__ xn_arr, const float* __restrict__ b_full,
    float* __restrict__ z_out, unsigned short* __restrict__ rh_bf,
    float* __restrict__ rhn_out, int H)
{
  extern __shared__ float sm[];
  int b = blockIdx.x, t = threadIdx.x;
  float hn = hn_arr[b], xn = xn_arr[b];
  float h_t = h[b * H + t];

  float z = gate_elem(Whz[b * ldWh + t], Ux[b * ldUx + 2 * H + t], b_full[2 * H + t], hn, xn, sm);
  float r = gate_elem(Whr[b * ldWh + t], Ux[b * ldUx + t],         b_full[t],         hn, xn, sm);
  z_out[b * H + t] = z;

  // rh = mobius_pointwise_mul(r, h)
  float wx  = r * h_t;
  float wxn = sqrtf(fmaxf(block_sum(wx * wx, sm), EPSF));
  float res = tanhf(wxn / hn * artanh_c(hn)) * wx / wxn;
  if (wxn <= 1e-7f) res = 0.0f;
  rh_bf[b * H + t] = f2bf(res);
  float rhn = sqrtf(fmaxf(block_sum(res * res, sm), EPSF));
  if (t == 0) rhn_out[b] = rhn;
}

__global__ __launch_bounds__(512) void gru_update_kernel(
    const float* __restrict__ Whh, const float* __restrict__ Ux, int ldUx,
    const float* __restrict__ z_in, const float* __restrict__ h,
    const float* __restrict__ rhn_arr, const float* __restrict__ xn_arr,
    const float* __restrict__ b_full,
    float* __restrict__ h_out, unsigned short* __restrict__ h_bf_out,
    float* __restrict__ hn_out, int H)
{
  extern __shared__ float sm[];
  int b = blockIdx.x, t = threadIdx.x;
  float xn = xn_arr[b], rhn = rhn_arr[b];
  float h_t = h[b * H + t], z_t = z_in[b * H + t];
  float whh = Whh[b * H + t];
  float uxh = Ux[b * ldUx + H + t];
  float bh  = b_full[H + t];

  // h_tilde = one_rnn_transform(W_hh_, rh, W_ih_, x, b_h)
  float whn = sqrtf(fmaxf(block_sum(whh * whh, sm), EPSF));
  float a   = mmv_scale(whh, whn, rhn);
  float uxn = sqrtf(fmaxf(block_sum(uxh * uxh, sm), EPSF));
  float bb  = mmv_scale(uxh, uxn, xn);
  float x2 = block_sum(a * a, sm);
  float y2 = block_sum(bb * bb, sm);
  float xy = block_sum(a * bb, sm);
  float c = mobius_add_elem(a, bb, x2, y2, xy);
  float c2 = block_sum(c * c, sm);
  float b2 = block_sum(bh * bh, sm);
  float cb = block_sum(c * bh, sm);
  float ht = mobius_add_elem(c, bh, c2, b2, cb);

  // delta = mobius_add(-h, h_tilde)
  float h2  = block_sum(h_t * h_t, sm);
  float ht2 = block_sum(ht * ht, sm);
  float hht = block_sum(-h_t * ht, sm);
  float delta = mobius_add_elem(-h_t, ht, h2, ht2, hht);

  // zd = mobius_pointwise_mul(z, delta)
  float dnrm = sqrtf(fmaxf(block_sum(delta * delta, sm), EPSF));
  float wx   = z_t * delta;
  float wxn  = sqrtf(fmaxf(block_sum(wx * wx, sm), EPSF));
  float zd   = tanhf(wxn / dnrm * artanh_c(dnrm)) * wx / wxn;
  if (wxn <= 1e-7f) zd = 0.0f;

  // h_new = mobius_add(h, zd)
  float zd2 = block_sum(zd * zd, sm);
  float hzd = block_sum(h_t * zd, sm);
  float hnew = mobius_add_elem(h_t, zd, h2, zd2, hzd);

  h_out[b * H + t] = hnew;
  h_bf_out[b * H + t] = f2bf(hnew);
  float hn_new = sqrtf(fmaxf(block_sum(hnew * hnew, sm), EPSF));
  if (t == 0) hn_out[b] = hn_new;
}

// ---------------------------------------------------------------------------
__global__ __launch_bounds__(512) void align_gather_kernel(
    const float* __restrict__ h_tgt, const float* __restrict__ hn_tgt,
    const int* __restrict__ alignment,
    float* __restrict__ h_al, unsigned short* __restrict__ h_al_bf,
    float* __restrict__ hn_al, int H)
{
  int b = blockIdx.x, t = threadIdx.x;
  int a = alignment[b];
  float v = h_tgt[a * H + t];
  h_al[b * H + t] = v;
  h_al_bf[b * H + t] = f2bf(v);
  if (t == 0) hn_al[b] = hn_tgt[a];
}

// One block per batch row; P=256 threads; H=512 handled 2 elems/thread.
__global__ __launch_bounds__(256) void final_logits_kernel(
    const float* __restrict__ Msrc, const float* __restrict__ Mtgt,
    const float* __restrict__ h_src, const float* __restrict__ hn_src,
    const float* __restrict__ h_tgt, const float* __restrict__ hn_tgt,
    const float* __restrict__ bp_src, const float* __restrict__ bp_tgt,
    const float* __restrict__ dist_bias,
    const float* __restrict__ plane_p, const float* __restrict__ plane_a,
    float* __restrict__ out, int H, int P, int Kp)
{
  extern __shared__ float sm[];
  int b = blockIdx.x, t = threadIdx.x;

  // d2 = poincare_dist(h_src, h_tgt)^2
  float x0 = h_src[b * H + t], x1 = h_src[b * H + t + P];
  float y0 = h_tgt[b * H + t], y1 = h_tgt[b * H + t + P];
  float x2 = block_sum(x0 * x0 + x1 * x1, sm);
  float y2 = block_sum(y0 * y0 + y1 * y1, sm);
  float xy = block_sum(-x0 * y0 - x1 * y1, sm);
  float d0 = mobius_add_elem(-x0, y0, x2, y2, xy);
  float d1 = mobius_add_elem(-x1, y1, x2, y2, xy);
  float dn = sqrtf(fmaxf(block_sum(d0 * d0 + d1 * d1, sm), EPSF));
  float dist = 2.0f * artanh_c(dn);
  float d2 = dist * dist;

  const float maxnorm = 1.0f - 1e-3f;

  // src_proj = project(mobius_add(mmv(Msrc), bp_src))
  float ms = Msrc[b * P + t];
  float msn = sqrtf(fmaxf(block_sum(ms * ms, sm), EPSF));
  float sp = mmv_scale(ms, msn, hn_src[b]);
  float bs = bp_src[t];
  float sp2 = block_sum(sp * sp, sm);
  float bs2 = block_sum(bs * bs, sm);
  float spbs = block_sum(sp * bs, sm);
  float sadd = mobius_add_elem(sp, bs, sp2, bs2, spbs);
  float snn = sqrtf(fmaxf(block_sum(sadd * sadd, sm), EPSF));
  float sproj = (snn > maxnorm) ? sadd / snn * maxnorm : sadd;

  // tgt_proj
  float mt = Mtgt[b * P + t];
  float mtn = sqrtf(fmaxf(block_sum(mt * mt, sm), EPSF));
  float tp = mmv_scale(mt, mtn, hn_tgt[b]);
  float bt = bp_tgt[t];
  float tp2 = block_sum(tp * tp, sm);
  float bt2 = block_sum(bt * bt, sm);
  float tpbt = block_sum(tp * bt, sm);
  float tadd = mobius_add_elem(tp, bt, tp2, bt2, tpbt);
  float tnn = sqrtf(fmaxf(block_sum(tadd * tadd, sm), EPSF));
  float tproj = (tnn > maxnorm) ? tadd / tnn * maxnorm : tadd;

  // projected = mobius_add(src_proj, tgt_proj)
  float s2 = block_sum(sproj * sproj, sm);
  float t2 = block_sum(tproj * tproj, sm);
  float st = block_sum(sproj * tproj, sm);
  float proj = mobius_add_elem(sproj, tproj, s2, t2, st);

  // bias = mobius_scalar_mul(d2, dist_bias)
  float db = dist_bias[t];
  float dbn = sqrtf(fmaxf(block_sum(db * db, sm), EPSF));
  float bias = tanhf(d2 * artanh_c(dbn)) * db / dbn;

  float p2 = block_sum(proj * proj, sm);
  float bias2 = block_sum(bias * bias, sm);
  float pb = block_sum(proj * bias, sm);
  float pr = mobius_add_elem(proj, bias, p2, bias2, pb);

  // dist2plane
  for (int k = 0; k < Kp; ++k) {
    float pk = plane_p[k * P + t];
    float ak = plane_a[k * P + t];
    float pk2 = block_sum(pk * pk, sm);
    float pr2 = block_sum(pr * pr, sm);
    float pkpr = block_sum(-pk * pr, sm);
    float diff = mobius_add_elem(-pk, pr, pk2, pr2, pkpr);
    float diff2 = block_sum(diff * diff, sm);
    float scda = block_sum(diff * ak, sm);
    float an = sqrtf(fmaxf(block_sum(ak * ak, sm), EPSF));
    float num = 2.0f * scda;
    float den = fmaxf((1.0f - diff2) * an, EPSF);
    float logit = asinhf(num / den);
    if (t == 0) out[b * Kp + k] = logit;
  }
}

// ---------------------------------------------------------------------------
extern "C" void kernel_launch(void* const* d_in, const int* in_sizes, int n_in,
                              void* d_out, int out_size, void* d_ws, size_t ws_size,
                              hipStream_t stream) {
  (void)in_sizes; (void)n_in; (void)out_size; (void)ws_size;
  const int E = 256, H = 512, P = 256, Kp = 2, B = 512, S = 100;

  const float* embedding = (const float*)d_in[0];
  const float* w_ih[2] = {(const float*)d_in[1], (const float*)d_in[4]};
  const float* w_hh[2] = {(const float*)d_in[2], (const float*)d_in[5]};
  const float* b_f[2]  = {(const float*)d_in[3], (const float*)d_in[6]};
  const float* wp_s = (const float*)d_in[7];
  const float* bp_s = (const float*)d_in[8];
  const float* wp_t = (const float*)d_in[9];
  const float* bp_t = (const float*)d_in[10];
  const float* dist_bias = (const float*)d_in[11];
  const float* plane_p = (const float*)d_in[12];
  const float* plane_a = (const float*)d_in[13];
  const int* tokens[2] = {(const int*)d_in[14], (const int*)d_in[15]};
  const int* alignment = (const int*)d_in[16];
  float* out = (float*)d_out;

  char* ws = (char*)d_ws;
  size_t off = 0;
  auto alloc = [&](size_t bytes) -> void* {
    void* p = ws + off;
    off = (off + bytes + 255) & ~(size_t)255;
    return p;
  };

  unsigned short* x_bf[2]; float* xn[2];
  unsigned short* wih_bf[2];
  unsigned short* whhzr_bf[2];  // fused [2H x H]: rows 0..H-1 = W_hz, H..2H-1 = W_hr
  unsigned short* whhh_bf[2];   // W_hh_ [H x H]
  float* hbuf[2]; unsigned short* hbf[2]; float* hn[2];
  for (int d = 0; d < 2; ++d) {
    x_bf[d]     = (unsigned short*)alloc((size_t)S * B * E * 2);
    xn[d]       = (float*)alloc((size_t)S * B * 4);
    wih_bf[d]   = (unsigned short*)alloc((size_t)3 * H * E * 2);
    whhzr_bf[d] = (unsigned short*)alloc((size_t)2 * H * H * 2);
    whhh_bf[d]  = (unsigned short*)alloc((size_t)H * H * 2);
    hbuf[d]     = (float*)alloc((size_t)B * H * 4);
    hbf[d]      = (unsigned short*)alloc((size_t)B * H * 2);
    hn[d]       = (float*)alloc((size_t)B * 4);
  }
  unsigned short* wp_bf[2] = {(unsigned short*)alloc((size_t)P * H * 2),
                              (unsigned short*)alloc((size_t)P * H * 2)};
  float* Whzr = (float*)alloc((size_t)B * 2 * H * 4);  // cols [0,H)=z, [H,2H)=r
  float* Whh  = (float*)alloc((size_t)B * H * 4);
  float* Ux   = (float*)alloc((size_t)B * 3 * H * 4);
  float* zbuf = (float*)alloc((size_t)B * H * 4);
  unsigned short* rh_bf = (unsigned short*)alloc((size_t)B * H * 2);
  float* rhn  = (float*)alloc((size_t)B * 4);
  float* h_al = (float*)alloc((size_t)B * H * 4);
  unsigned short* h_al_bf = (unsigned short*)alloc((size_t)B * H * 2);
  float* hn_al = (float*)alloc((size_t)B * 4);
  float* Msrc = (float*)alloc((size_t)B * P * 4);
  float* Mtgt = (float*)alloc((size_t)B * P * 4);

  auto conv = [&](const float* s, unsigned short* dst, int n) {
    f32_to_bf16_kernel<<<(n + 255) / 256, 256, 0, stream>>>(s, dst, n);
  };
  for (int d = 0; d < 2; ++d) {
    conv(w_ih[d], wih_bf[d], 3 * H * E);
    conv(w_hh[d] + (size_t)2 * H * H, whhzr_bf[d],                 H * H); // z first
    conv(w_hh[d],                     whhzr_bf[d] + (size_t)H * H, H * H); // then r
    conv(w_hh[d] + (size_t)H * H,     whhh_bf[d],                  H * H); // W_hh_
  }
  conv(wp_s, wp_bf[0], P * H);
  conv(wp_t, wp_bf[1], P * H);

  for (int d = 0; d < 2; ++d) {
    embed_expmap0_kernel<<<S * B, 256, 256 * 4, stream>>>(
        embedding, tokens[d], x_bf[d], xn[d], B, S, E);
    init_h_kernel<<<(B * H + 255) / 256, 256, 0, stream>>>(hbuf[d], hbf[d], hn[d], B * H, B);
  }

  for (int d = 0; d < 2; ++d) {
    for (int s = 0; s < S; ++s) {
      const unsigned short* xt = x_bf[d] + (size_t)s * B * E;
      const float* xnt = xn[d] + (size_t)s * B;
      // Ux = x_t @ w_ih^T  [B, 3H]  (cols: [0,H)=r, [H,2H)=h, [2H,3H)=z)
      wmma_gemm_kernel<<<dim3(3 * H / 64, B / 64), 128, 0, stream>>>(
          xt, E, wih_bf[d], E, Ux, 3 * H, E);
      // Whzr = h @ [W_hz; W_hr]^T  [B, 2H]
      wmma_gemm_kernel<<<dim3(2 * H / 64, B / 64), 128, 0, stream>>>(
          hbf[d], H, whhzr_bf[d], H, Whzr, 2 * H, H);
      gru_gates_kernel<<<B, H, H * 4, stream>>>(
          Whzr, Whzr + H, 2 * H, Ux, 3 * H, hbuf[d], hn[d], xnt, b_f[d],
          zbuf, rh_bf, rhn, H);
      // Whh = rh @ W_hh_^T
      wmma_gemm_kernel<<<dim3(H / 64, B / 64), 128, 0, stream>>>(
          rh_bf, H, whhh_bf[d], H, Whh, H, H);
      gru_update_kernel<<<B, H, H * 4, stream>>>(
          Whh, Ux, 3 * H, zbuf, hbuf[d], rhn, xnt, b_f[d], hbuf[d], hbf[d], hn[d], H);
    }
  }

  align_gather_kernel<<<B, H, 0, stream>>>(hbuf[1], hn[1], alignment,
                                           h_al, h_al_bf, hn_al, H);
  wmma_gemm_kernel<<<dim3(P / 64, B / 64), 128, 0, stream>>>(
      hbf[0], H, wp_bf[0], H, Msrc, P, H);
  wmma_gemm_kernel<<<dim3(P / 64, B / 64), 128, 0, stream>>>(
      h_al_bf, H, wp_bf[1], H, Mtgt, P, H);
  final_logits_kernel<<<B, P, P * 4, stream>>>(
      Msrc, Mtgt, hbuf[0], hn[0], h_al, hn_al,
      bp_s, bp_t, dist_bias, plane_p, plane_a, out, H, P, Kp);
}
